// AssignmentSimilarityNet_62122406969462
// MI455X (gfx1250) — matrix operand order repaired and checked by reading, not compiled, not run on hardware
//
#include <hip/hip_runtime.h>
#include <hip/hip_bf16.h>
#include <cstdint>

typedef __attribute__((ext_vector_type(16))) _Float16 v16h;
typedef __attribute__((ext_vector_type(8)))  float    v8f;

#define BM 128
#define BN 128
#define BK 32
#define LDST 40  // padded LDS row stride (halves): 80B rows, 16B-aligned, conflict-spread

union F16x16 { uint4 u[2]; v16h v; };

__device__ __forceinline__ void async_copy_b128(unsigned lds_off, const void* gptr)
{
    // GLOBAL_LOAD_ASYNC_TO_LDS_B128: per-lane 16B global -> LDS, tracked by ASYNCcnt.
    asm volatile("global_load_async_to_lds_b128 %0, %1, off"
                 :: "v"(lds_off), "v"((unsigned long long)(uintptr_t)gptr)
                 : "memory");
}
__device__ __forceinline__ void wait_async0()
{
    asm volatile("s_wait_asynccnt 0x0" ::: "memory");
}

// ---------------------------------------------------------------------------
// Generic fused WMMA GEMM:  out = epi( A[M,K] @ B[K,N] + bias + T1[a]+C1[b] )
// A f16 row-major (M,K). B passed TRANSPOSED: BT f16 row-major (N,K), leading
// dim ldbt. Optional second A (DUAL concat along K at Ksplit).
// Requires M%128==0, N%128==0, K%32==0 (true for all calls here).
// ---------------------------------------------------------------------------
__global__ __launch_bounds__(256) void wmma_gemm_k(
    const _Float16* __restrict__ A, const _Float16* __restrict__ A2, int Ksplit,
    const _Float16* __restrict__ BT,
    const float* __restrict__ bias,
    const float* __restrict__ rowA, const float* __restrict__ rowB, int rowDiv,
    _Float16* __restrict__ outH, float* __restrict__ outF,
    int M, int N, int K, int lda, int ldbt, int ldc, int ocs,
    int relu, int oneMinus)
{
    __shared__ _Float16 As[2][BM][LDST];
    __shared__ _Float16 Bs[2][BN][LDST];   // [n][k]

    const int tid   = threadIdx.x;
    const int lane  = tid & 31;
    const int wave  = tid >> 5;
    const int waveM = wave & 3;            // 4 wave-rows  x 32
    const int waveN = wave >> 2;           // 2 wave-cols  x 64
    const int bM = blockIdx.y * BM;
    const int bN = blockIdx.x * BN;
    const int nk = K / BK;

    // chunk geometry (identical for A and B tiles): 128 rows x 32 halves
    // = 512 chunks of 8 halves; thread handles chunks tid and tid+256.
    const int r0  = tid >> 2;              // 0..63
    const int c80 = (tid & 3) << 3;        // 0,8,16,24
    const int r1  = (tid + 256) >> 2;      // 64..127

    v8f acc[2][4];
#pragma unroll
    for (int mi = 0; mi < 2; ++mi)
#pragma unroll
        for (int ni = 0; ni < 4; ++ni)
#pragma unroll
            for (int r = 0; r < 8; ++r) acc[mi][ni][r] = 0.0f;

    // direct memory -> LDS async copies (no staging VGPRs, ASYNCcnt tracked)
    auto issue_async = [&](int kt, int buf) {
        const int k0 = kt * BK;
        const _Float16* Ap = A;
        int ak = k0;
        if (A2 != nullptr && k0 >= Ksplit) { Ap = A2; ak = k0 - Ksplit; }
        async_copy_b128((unsigned)(uintptr_t)&As[buf][r0][c80],
                        Ap + (size_t)(bM + r0) * lda + ak + c80);
        async_copy_b128((unsigned)(uintptr_t)&As[buf][r1][c80],
                        Ap + (size_t)(bM + r1) * lda + ak + c80);
        async_copy_b128((unsigned)(uintptr_t)&Bs[buf][r0][c80],
                        BT + (size_t)(bN + r0) * ldbt + k0 + c80);
        async_copy_b128((unsigned)(uintptr_t)&Bs[buf][r1][c80],
                        BT + (size_t)(bN + r1) * ldbt + k0 + c80);
    };

    issue_async(0, 0);

    const int amr = lane & 15;             // A fragment row (M)
    const int ako = (lane >> 4) << 3;      // K sub-offset 0/8 (A layout)
    const int bnc = lane & 15;             // B fragment col (N)
    const int bko = (lane >> 4) << 3;      // K sub-offset 0/8 (B layout)

    for (int kt = 0; kt < nk; ++kt) {
        const int cur = kt & 1;
        wait_async0();        // my copies for buffer `cur` have landed in LDS
        __syncthreads();      // everyone's copies done; everyone past reads of cur^1
        if (kt + 1 < nk) issue_async(kt + 1, cur ^ 1);   // DMA overlaps WMMA below

        v16h af[2], bf[4];
#pragma unroll
        for (int mi = 0; mi < 2; ++mi) {
            const int r = waveM * 32 + mi * 16 + amr;
            F16x16 u;
            u.u[0] = *(const uint4*)&As[cur][r][ako];        // K: ako..ako+7
            u.u[1] = *(const uint4*)&As[cur][r][16 + ako];   // K: 16+ako..
            af[mi] = u.v;
        }
#pragma unroll
        for (int ni = 0; ni < 4; ++ni) {
            const int cidx = waveN * 64 + ni * 16 + bnc;
            F16x16 u;
            u.u[0] = *(const uint4*)&Bs[cur][cidx][bko];     // K: bko..bko+7
            u.u[1] = *(const uint4*)&Bs[cur][cidx][16 + bko]; // K: 16+bko..16+bko+7
            bf[ni] = u.v;
        }
#pragma unroll
        for (int mi = 0; mi < 2; ++mi)
#pragma unroll
            for (int ni = 0; ni < 4; ++ni)
                acc[mi][ni] = __builtin_amdgcn_wmma_f32_16x16x32_f16(
                    false, af[mi], false, bf[ni], (short)0, acc[mi][ni], false, false);
    }

    // Epilogue. C/D layout: lanes 0-15 -> M=r, N=lane; lanes 16-31 -> M=8+r.
#pragma unroll
    for (int mi = 0; mi < 2; ++mi) {
#pragma unroll
        for (int ni = 0; ni < 4; ++ni) {
            const int gcol  = bN + waveN * 64 + ni * 16 + (lane & 15);
            const int rbase = bM + waveM * 32 + mi * 16 + ((lane >> 4) << 3);
            const float bv  = bias ? bias[gcol] : 0.0f;
#pragma unroll
            for (int r = 0; r < 8; ++r) {
                const int grow = rbase + r;
                float v = acc[mi][ni][r] + bv;
                if (rowA != nullptr) {
                    const int ga = grow / rowDiv;
                    const int gb = grow - ga * rowDiv;
                    v += rowA[(size_t)ga * N + gcol] + rowB[(size_t)gb * N + gcol];
                }
                if (oneMinus) v = 1.0f - v;
                if (relu)     v = fmaxf(v, 0.0f);
                const size_t oi = (size_t)grow * (size_t)ldc + (size_t)gcol * (size_t)ocs;
                if (outH) outH[oi] = (_Float16)v;
                if (outF) outF[oi] = v;
            }
        }
    }
}

// ---------------------------------------------------------------------------
// Row L2-normalize + f16 convert (normalized row-major + raw copy).
// Normalized current_app doubles as BT for the distance GEMM.
// ---------------------------------------------------------------------------
__global__ void norm_app_k(const float* __restrict__ X,
                           _Float16* __restrict__ Xn, _Float16* __restrict__ Xr)
{
    __shared__ float red[256];
    const int row = blockIdx.x, t = threadIdx.x;
    const float x0 = X[row * 512 + t];
    const float x1 = X[row * 512 + 256 + t];
    red[t] = x0 * x0 + x1 * x1;
    __syncthreads();
    for (int s = 128; s > 0; s >>= 1) { if (t < s) red[t] += red[t + s]; __syncthreads(); }
    const float rn = rsqrtf(red[0]);
    Xr[row * 512 + t]       = (_Float16)x0;
    Xr[row * 512 + 256 + t] = (_Float16)x1;
    Xn[row * 512 + t]       = (_Float16)(x0 * rn);
    Xn[row * 512 + 256 + t] = (_Float16)(x1 * rn);
}

// ---------------------------------------------------------------------------
// Geometric edge features into padded (AB x 32) f16 tensor (col 5 = dist,
// filled later by the distance GEMM's strided epilogue; cols 6..31 = 0).
// ---------------------------------------------------------------------------
__global__ void edge_geom_k(const float* __restrict__ tc, const float* __restrict__ cc,
                            const float* __restrict__ tt, const float* __restrict__ ct,
                            _Float16* __restrict__ ef)
{
    const int idx = blockIdx.x * 256 + threadIdx.x;   // 576*256 == 147456 exactly
    const int a = idx / 384;
    const int b = idx - a * 384;
    const float tx1 = tc[a * 4 + 0], ty1 = tc[a * 4 + 1], tx2 = tc[a * 4 + 2], ty2 = tc[a * 4 + 3];
    const float cx1 = cc[b * 4 + 0], cy1 = cc[b * 4 + 1], cx2 = cc[b * 4 + 2], cy2 = cc[b * 4 + 3];
    const float trx = (tx1 + tx2) * 0.5f, tryy = (ty1 + ty2) * 0.5f;
    const float trw = fabsf(tx2 - tx1),  trh  = fabsf(ty1 - ty2);
    const float cux = (cx1 + cx2) * 0.5f, cuy = (cy1 + cy2) * 0.5f;
    const float cuw = fabsf(cx2 - cx1),  cuh  = fabsf(cy1 - cy2);
    const float inv = 1.0f / (trw + cuw);
    union { _Float16 h[32]; uint4 u[4]; } v;
#pragma unroll
    for (int j = 0; j < 32; ++j) v.h[j] = (_Float16)0.0f;
    v.h[0] = (_Float16)(2.0f * (cux - trx) * inv);
    v.h[1] = (_Float16)(2.0f * (cuy - tryy) * inv);
    v.h[2] = (_Float16)logf(trw / cuw);
    v.h[3] = (_Float16)logf(trh / cuh);
    v.h[4] = (_Float16)(tt[a] - ct[b]);
    uint4* dst = (uint4*)(ef + (size_t)idx * 32);
    dst[0] = v.u[0]; dst[1] = v.u[1]; dst[2] = v.u[2]; dst[3] = v.u[3];
}

// ---------------------------------------------------------------------------
// Edge sums -> node-MLP input concat buffers
// ---------------------------------------------------------------------------
__global__ void rowsum_a_k(const _Float16* __restrict__ E, const _Float16* __restrict__ tn,
                           _Float16* __restrict__ na)
{
    const int a = blockIdx.x, j = threadIdx.x;     // 128 threads
    const _Float16* base = E + (size_t)a * 384 * 128 + j;
    float s = 0.0f;
    for (int b = 0; b < 384; ++b) s += (float)base[(size_t)b * 128];
    na[a * 256 + j]       = tn[a * 128 + j];
    na[a * 256 + 128 + j] = (_Float16)s;
}

__global__ void rowsum_b_k(const _Float16* __restrict__ E, const _Float16* __restrict__ cn,
                           _Float16* __restrict__ nb)
{
    const int b = blockIdx.x, j = threadIdx.x;
    const _Float16* base = E + (size_t)b * 128 + j;
    float s = 0.0f;
    for (int a = 0; a < 384; ++a) s += (float)base[(size_t)a * 384 * 128];
    nb[b * 256 + j]       = cn[b * 128 + j];
    nb[b * 256 + 128 + j] = (_Float16)s;
}

// Classifier second layer: (AB,128) @ (128,1) + b
__global__ void cls_matvec_k(const _Float16* __restrict__ G, const _Float16* __restrict__ w2,
                             const float* __restrict__ b2, float* __restrict__ out)
{
    __shared__ float sw[128];
    const int t = threadIdx.x;
    if (t < 128) sw[t] = (float)w2[t];
    __syncthreads();
    const int m = blockIdx.x * 256 + t;            // 576*256 == 147456 exactly
    const _Float16* g = G + (size_t)m * 128;
    float s = b2[0];
#pragma unroll 16
    for (int j = 0; j < 128; ++j) s += (float)g[j] * sw[j];
    out[m] = s;
}

// f32 (K,N) -> f16 transposed (N,K): d[n*K+k] = s[k*N+n]
__global__ void f32_to_f16_T_k(const float* __restrict__ s, _Float16* __restrict__ d,
                               int K, int N)
{
    const int i = blockIdx.x * 256 + threadIdx.x;
    if (i < K * N) {
        const int k = i / N;
        const int n = i - k * N;
        d[(size_t)n * K + k] = (_Float16)s[i];
    }
}

__global__ void f32_to_f16_k(const float* __restrict__ s, _Float16* __restrict__ d, int n)
{
    const int i = blockIdx.x * 256 + threadIdx.x;
    if (i < n) d[i] = (_Float16)s[i];
}

// W_ei1 (6,128) -> transposed zero-padded (128,32) f16
__global__ void pad_ei1_T_k(const float* __restrict__ s, _Float16* __restrict__ d)
{
    const int i = blockIdx.x * 256 + threadIdx.x;
    if (i < 128 * 32) {
        const int n = i >> 5;
        const int k = i & 31;
        d[i] = (k < 6) ? (_Float16)s[k * 128 + n] : (_Float16)0.0f;
    }
}

// ---------------------------------------------------------------------------
// Host orchestration
// ---------------------------------------------------------------------------
static void launch_gemm(hipStream_t st,
    const _Float16* A, const _Float16* A2, int Ksplit, const _Float16* BT,
    const float* bias, const float* rA, const float* rB, int rowDiv,
    _Float16* outH, float* outF,
    int M, int N, int K, int lda, int ldbt, int ldc, int ocs, int relu, int onem)
{
    dim3 grid(N / BN, M / BM);
    wmma_gemm_k<<<grid, 256, 0, st>>>(A, A2, Ksplit, BT, bias, rA, rB, rowDiv,
                                      outH, outF, M, N, K, lda, ldbt, ldc, ocs, relu, onem);
}

extern "C" void kernel_launch(void* const* d_in, const int* in_sizes, int n_in,
                              void* d_out, int out_size, void* d_ws, size_t ws_size,
                              hipStream_t stream)
{
    (void)in_sizes; (void)n_in; (void)out_size; (void)ws_size;
    const float* track_app     = (const float*)d_in[0];
    const float* current_app   = (const float*)d_in[1];
    const float* track_coords  = (const float*)d_in[2];
    const float* current_coords= (const float*)d_in[3];
    const float* track_t       = (const float*)d_in[4];
    const float* curr_t        = (const float*)d_in[5];
    const float* W_cnn = (const float*)d_in[6];  const float* b_cnn = (const float*)d_in[7];
    const float* W_ei1 = (const float*)d_in[8];  const float* b_ei1 = (const float*)d_in[9];
    const float* W_ei2 = (const float*)d_in[10]; const float* b_ei2 = (const float*)d_in[11];
    const float* W_e1  = (const float*)d_in[12]; const float* b_e1  = (const float*)d_in[13];
    const float* W_e2  = (const float*)d_in[14]; const float* b_e2  = (const float*)d_in[15];
    const float* W_n1  = (const float*)d_in[16]; const float* b_n1  = (const float*)d_in[17];
    const float* W_n2  = (const float*)d_in[18]; const float* b_n2  = (const float*)d_in[19];
    const float* W_c1  = (const float*)d_in[20]; const float* b_c1  = (const float*)d_in[21];
    const float* W_c2  = (const float*)d_in[22]; const float* b_c2  = (const float*)d_in[23];
    float* out = (float*)d_out;

    const int AB = 384 * 384;   // 147456

    // workspace carve-out (256B aligned)
    size_t off = 0;
    auto alloc = [&](size_t bytes) -> char* {
        size_t p = (off + 255) & ~(size_t)255;
        off = p + bytes;
        return (char*)d_ws + p;
    };
    _Float16* taN   = (_Float16*)alloc(384 * 512 * 2);  // normalized track (A operand)
    _Float16* caN   = (_Float16*)alloc(384 * 512 * 2);  // normalized current (BT operand!)
    _Float16* taR   = (_Float16*)alloc(384 * 512 * 2);
    _Float16* caR   = (_Float16*)alloc(384 * 512 * 2);
    _Float16* wcnnT = (_Float16*)alloc(128 * 512 * 2);
    _Float16* wei1T = (_Float16*)alloc(128 * 32 * 2);
    _Float16* wei2T = (_Float16*)alloc(128 * 128 * 2);
    _Float16* we1T  = (_Float16*)alloc(128 * 512 * 2);  // (N=128, K=512)
    _Float16* we2T  = (_Float16*)alloc(128 * 128 * 2);
    _Float16* wn1T  = (_Float16*)alloc(128 * 256 * 2);
    _Float16* wn2T  = (_Float16*)alloc(128 * 128 * 2);
    _Float16* wc1T  = (_Float16*)alloc(128 * 128 * 2);
    _Float16* wc2   = (_Float16*)alloc(128 * 2);
    _Float16* ef    = (_Float16*)alloc((size_t)AB * 32 * 2);
    _Float16* e0    = (_Float16*)alloc((size_t)AB * 128 * 2);
    _Float16* eb    = (_Float16*)alloc((size_t)AB * 128 * 2);
    _Float16* hb    = (_Float16*)alloc((size_t)AB * 128 * 2);
    float*    T1    = (float*)   alloc(384 * 128 * 4);
    float*    C1    = (float*)   alloc(384 * 128 * 4);
    _Float16* tn    = (_Float16*)alloc(384 * 128 * 2);
    _Float16* cn    = (_Float16*)alloc(384 * 128 * 2);
    _Float16* na    = (_Float16*)alloc(384 * 256 * 2);
    _Float16* nb    = (_Float16*)alloc(384 * 256 * 2);
    _Float16* nth   = (_Float16*)alloc(384 * 128 * 2);

    // weight conversions (transposed: BT layout, done once per call)
    f32_to_f16_T_k<<<(512*128+255)/256, 256, 0, stream>>>(W_cnn, wcnnT, 512, 128);
    pad_ei1_T_k   <<<(128*32+255)/256,  256, 0, stream>>>(W_ei1, wei1T);
    f32_to_f16_T_k<<<(128*128+255)/256, 256, 0, stream>>>(W_ei2, wei2T, 128, 128);
    f32_to_f16_T_k<<<(512*128+255)/256, 256, 0, stream>>>(W_e1,  we1T,  512, 128);
    f32_to_f16_T_k<<<(128*128+255)/256, 256, 0, stream>>>(W_e2,  we2T,  128, 128);
    f32_to_f16_T_k<<<(256*128+255)/256, 256, 0, stream>>>(W_n1,  wn1T,  256, 128);
    f32_to_f16_T_k<<<(128*128+255)/256, 256, 0, stream>>>(W_n2,  wn2T,  128, 128);
    f32_to_f16_T_k<<<(128*128+255)/256, 256, 0, stream>>>(W_c1,  wc1T,  128, 128);
    f32_to_f16_k  <<<1, 256, 0, stream>>>(W_c2, wc2, 128);

    // normalize appearances (+ raw f16 copies). caN is already the BT operand.
    norm_app_k<<<384, 256, 0, stream>>>(track_app,   taN, taR);
    norm_app_k<<<384, 256, 0, stream>>>(current_app, caN, caR);

    // geometric edge features (cols 0..4, zeros elsewhere)
    edge_geom_k<<<576, 256, 0, stream>>>(track_coords, current_coords, track_t, curr_t, ef);

    // dist_reid = 1 - taN @ caN^T ; write strided into ef column 5
    launch_gemm(stream, taN, nullptr, 0, caN, nullptr, nullptr, nullptr, 1,
                ef + 5, nullptr, 384, 384, 512, 512, 512, /*ldc*/384*32, /*ocs*/32,
                /*relu*/0, /*onem*/1);

    // edge_in MLP:  hb = relu(ef @ Wei1 + b_ei1);  e0 = relu(hb @ Wei2 + b_ei2)
    launch_gemm(stream, ef, nullptr, 0, wei1T, b_ei1, nullptr, nullptr, 1,
                hb, nullptr, AB, 128, 32, 32, 32, 128, 1, 1, 0);
    launch_gemm(stream, hb, nullptr, 0, wei2T, b_ei2, nullptr, nullptr, 1,
                e0, nullptr, AB, 128, 128, 128, 128, 128, 1, 1, 0);

    // node embeddings from raw appearances
    launch_gemm(stream, taR, nullptr, 0, wcnnT, b_cnn, nullptr, nullptr, 1,
                tn, nullptr, 384, 128, 512, 512, 512, 128, 1, 1, 0);
    launch_gemm(stream, caR, nullptr, 0, wcnnT, b_cnn, nullptr, nullptr, 1,
                cn, nullptr, 384, 128, 512, 512, 512, 128, 1, 1, 0);

    const _Float16* e_in = e0;
    for (int s = 0; s < 4; ++s) {
        // hoisted node contributions of W_e1 (K cols 256..383 -> tracks, 384..511 -> currents)
        launch_gemm(stream, tn, nullptr, 0, we1T + 256, nullptr, nullptr, nullptr, 1,
                    nullptr, T1, 384, 128, 128, 128, 512, 128, 1, 0, 0);
        launch_gemm(stream, cn, nullptr, 0, we1T + 384, nullptr, nullptr, nullptr, 1,
                    nullptr, C1, 384, 128, 128, 128, 512, 128, 1, 0, 0);
        // hb = relu([e_in | e0] @ We1[0:256] + T1[a] + C1[b] + b_e1)
        launch_gemm(stream, e_in, e0, 128, we1T, b_e1, T1, C1, 384,
                    hb, nullptr, AB, 128, 256, 128, 512, 128, 1, 1, 0);
        // eb = relu(hb @ We2 + b_e2)   (new edge embeds)
        launch_gemm(stream, hb, nullptr, 0, we2T, b_e2, nullptr, nullptr, 1,
                    eb, nullptr, AB, 128, 128, 128, 128, 128, 1, 1, 0);
        // edge sums + concat with current node embeds
        rowsum_a_k<<<384, 128, 0, stream>>>(eb, tn, na);
        rowsum_b_k<<<384, 128, 0, stream>>>(eb, cn, nb);
        // node MLPs (in-place update of tn/cn)
        launch_gemm(stream, na, nullptr, 0, wn1T, b_n1, nullptr, nullptr, 1,
                    nth, nullptr, 384, 128, 256, 256, 256, 128, 1, 1, 0);
        launch_gemm(stream, nth, nullptr, 0, wn2T, b_n2, nullptr, nullptr, 1,
                    tn, nullptr, 384, 128, 128, 128, 128, 128, 1, 1, 0);
        launch_gemm(stream, nb, nullptr, 0, wn1T, b_n1, nullptr, nullptr, 1,
                    nth, nullptr, 384, 128, 256, 256, 256, 128, 1, 1, 0);
        launch_gemm(stream, nth, nullptr, 0, wn2T, b_n2, nullptr, nullptr, 1,
                    cn, nullptr, 384, 128, 128, 128, 128, 128, 1, 1, 0);
        // classifier: hb = relu(eb @ Wc1 + b_c1); out[s] = hb @ Wc2 + b_c2
        launch_gemm(stream, eb, nullptr, 0, wc1T, b_c1, nullptr, nullptr, 1,
                    hb, nullptr, AB, 128, 128, 128, 128, 128, 1, 1, 0);
        cls_matvec_k<<<576, 256, 0, stream>>>(hb, wc2, b_c2, out + (size_t)s * AB);
        e_in = eb;  // steps 1..3 update eb in place (row-local GEMM is safe)
    }
}
